// RPN_52243982189267
// MI455X (gfx1250) — matrix operand order repaired and checked by reading, not compiled, not run on hardware
//
#include <hip/hip_runtime.h>
#include <hip/hip_bf16.h>

typedef unsigned short u16b;
typedef __attribute__((ext_vector_type(16))) __bf16 v16bf;
typedef __attribute__((ext_vector_type(8)))  float  v8f;

#define BATCH 4
#define CIN   256
#define KTOT  2304          // 9 * 256 (im2col K, ordered k = tap*256 + c)
#define NPIX_TOT 8400       // 6400 + 1600 + 400
#define NANCH 25200         // 3 * 8400
#define SORTN 32768
#define TOPK  3000
#define IOU_THR 0.7f
#define SCALE_CLAMPF 4.135166556742356f  // log(1000/16)

__device__ __forceinline__ u16b f2bf(float x) {
  unsigned u = __float_as_uint(x);
  u += 0x7FFFu + ((u >> 16) & 1u);     // round-to-nearest-even
  return (u16b)(u >> 16);
}

union Frag { v16bf v; uint4 q[2]; };

// ---------------------------------------------------------------- repack ----
__global__ void k_nhwc_bf16(const float* __restrict__ src, u16b* __restrict__ dst,
                            int HW, int pixoff) {
  int id = blockIdx.x * blockDim.x + threadIdx.x;
  if (id >= BATCH * HW * CIN) return;
  int c = id & 255;
  int bp = id >> 8;
  int p = bp % HW;
  int b = bp / HW;
  float v = src[((size_t)(b * CIN + c)) * HW + p];
  dst[((size_t)(b * NPIX_TOT + pixoff + p)) * CIN + c] = f2bf(v);
}

// OIHW f32 -> [N][K] bf16 with k = tap*256 + c; rows beyond NREAL are zero.
__global__ void k_prep_w(const float* __restrict__ w, u16b* __restrict__ dst,
                         int NOUT, int NREAL) {
  int id = blockIdx.x * blockDim.x + threadIdx.x;
  if (id >= NOUT * KTOT) return;
  int k = id % KTOT, n = id / KTOT;
  int t = k >> 8, c = k & 255;
  float v = (n < NREAL) ? w[((size_t)(n * CIN + c)) * 9 + t] : 0.0f;
  dst[id] = f2bf(v);
}

__global__ void k_zero16(u16b* __restrict__ p, int n) {
  int i = blockIdx.x * blockDim.x + threadIdx.x;
  if (i < n) p[i] = 0;
}

// ------------------------------------------------- conv1: 256->256 + ReLU ---
// Implicit GEMM: M = H*W, N = 256, K = 2304. Tile 128x128, 8 waves, wave tile
// 64x32 = 8 x (16x16x32 bf16 WMMA) per K-step. A/B tiles staged with
// GLOBAL_LOAD_ASYNC_TO_LDS_B128 into double-buffered LDS; out-of-bounds halo
// lanes are redirected to a zeroed scratch page so no zero-fill VALU is needed.
__global__ __launch_bounds__(256) void k_conv1(
    const u16b* __restrict__ fb,   // NHWC bf16, this scale (batch stride 8400*256)
    const u16b* __restrict__ wB,   // [256][2304]
    const float* __restrict__ bias,
    u16b* __restrict__ y1,         // NHWC bf16 out, this scale
    const u16b* __restrict__ zp,   // >=32B of zeros
    int H, int W) {
  __shared__ __align__(16) u16b As[2][128 * 32];
  __shared__ __align__(16) u16b Bs[2][128 * 32];

  const int HW = H * W;
  const int tid  = threadIdx.x;
  const int lane = tid & 31;
  const int wid  = tid >> 5;
  const int wm = wid & 1, wn = wid >> 1;           // 2 waves on M, 4 on N
  const int tileM = blockIdx.x * 128;
  const int tileN = blockIdx.y * 128;
  const int b = blockIdx.z;
  const size_t bstr = (size_t)NPIX_TOT * CIN;
  const u16b* fbb = fb + (size_t)b * bstr;

  v8f acc[4][2];
  for (int i = 0; i < 4; ++i)
    for (int j = 0; j < 2; ++j)
      for (int e = 0; e < 8; ++e) acc[i][j][e] = 0.0f;

  const int r = tid >> 1, half = tid & 1;          // 2 threads per 32-elem row
  const int m = tileM + r;
  const int py = (m < HW) ? (m / W) : 0;
  const int px = (m < HW) ? (m % W) : 0;
  const u16b* wBrow = wB + (size_t)(tileN + r) * KTOT + half * 16;

  // low 32 bits of a flat LDS pointer == byte offset from LDS base
  const unsigned asBase = (unsigned)(size_t)(const void*)&As[0][0];
  const unsigned bsBase = (unsigned)(size_t)(const void*)&Bs[0][0];
  const unsigned thrOff = (unsigned)((r * 32 + half * 16) * 2);

  // Issue the async staging of K-step `s` into LDS buffer `bufb`.
  // offset:16 applies to BOTH the LDS and global address (ISA 08 §4.4).
  auto issue_stage = [&](int s, int bufb) {
    const int tap = s >> 3;                 // 8 c-steps of 32 per tap
    const int dy = tap / 3 - 1, dx = tap % 3 - 1;
    const int yy = py + dy, xx = px + dx;
    const bool ok = (m < HW) & (yy >= 0) & (yy < H) & (xx >= 0) & (xx < W);
    const u16b* gA = ok ? (fbb + (size_t)(yy * W + xx) * CIN + ((s & 7) << 5) + half * 16)
                        : zp;
    const u16b* gB = wBrow + (s << 5);
    const unsigned aO = asBase + (unsigned)bufb * 8192u + thrOff;
    const unsigned bO = bsBase + (unsigned)bufb * 8192u + thrOff;
    asm volatile(
        "global_load_async_to_lds_b128 %0, %2, off\n\t"
        "global_load_async_to_lds_b128 %0, %2, off offset:16\n\t"
        "global_load_async_to_lds_b128 %1, %3, off\n\t"
        "global_load_async_to_lds_b128 %1, %3, off offset:16"
        :: "v"(aO), "v"(bO), "v"(gA), "v"(gB)
        : "memory");
  };

  issue_stage(0, 0);

  for (int kt = 0; kt < 72; ++kt) {
    const int buf = kt & 1;
    // own share of stage kt has landed in LDS
    asm volatile("s_wait_asynccnt 0" ::: "memory");
    // everyone's share landed; everyone done reading buf^1 from stage kt-1
    __syncthreads();
    if (kt + 1 < 72) issue_stage(kt + 1, buf ^ 1);

    // A fragment (16x32 bf16): lanes 0-15 K{0..7,16..23}, lanes 16-31 K{8..15,24..31}
    const int khalfA = (lane >> 4) * 8;
    const int rowL   = lane & 15;
    Frag af[4];
    for (int mi = 0; mi < 4; ++mi) {
      const int R = wm * 64 + mi * 16 + rowL;
      af[mi].q[0] = *(const uint4*)&As[buf][R * 32 + khalfA];
      af[mi].q[1] = *(const uint4*)&As[buf][R * 32 + 16 + khalfA];
    }
    // B fragment (32x16 bf16), Bs is [n][k]: lanes 0-15 K{0..7,16..23}, lanes 16-31 K{8..15,24..31}
    const int kbB = (lane >> 4) * 8;
    Frag bfv[2];
    for (int ni = 0; ni < 2; ++ni) {
      const int N = wn * 32 + ni * 16 + rowL;
      bfv[ni].q[0] = *(const uint4*)&Bs[buf][N * 32 + kbB];
      bfv[ni].q[1] = *(const uint4*)&Bs[buf][N * 32 + kbB + 16];
    }
    for (int mi = 0; mi < 4; ++mi)
      for (int ni = 0; ni < 2; ++ni)
        acc[mi][ni] = __builtin_amdgcn_wmma_f32_16x16x32_bf16(
            false, af[mi].v, false, bfv[ni].v, (short)0, acc[mi][ni], false, false);
  }

  // epilogue: bias + ReLU -> NHWC bf16
  const int lhalf = lane >> 4, lmod = lane & 15;
  for (int mi = 0; mi < 4; ++mi)
    for (int ni = 0; ni < 2; ++ni) {
      const int ncol = tileN + wn * 32 + ni * 16 + lmod;
      const float bv = bias[ncol];
      for (int v = 0; v < 8; ++v) {
        const int mrow = tileM + wm * 64 + mi * 16 + v + lhalf * 8;
        if (mrow < HW) {
          float val = acc[mi][ni][v] + bv;
          val = val > 0.0f ? val : 0.0f;
          y1[(size_t)b * bstr + (size_t)mrow * CIN + ncol] = f2bf(val);
        }
      }
    }
}

// --------------------------------------------------- conv2: 256->15 (pad16) -
__global__ __launch_bounds__(256) void k_conv2(
    const u16b* __restrict__ y1,   // NHWC bf16, this scale
    const u16b* __restrict__ wB,   // [16][2304]
    const float* __restrict__ bias,// 15
    float* __restrict__ scores,    // [B][25200]
    float* __restrict__ deltas,    // [B][25200][4]
    int H, int W, int aoff) {
  __shared__ __align__(16) u16b As[128 * 32];
  __shared__ __align__(16) u16b Bs[16 * 32];

  const int HW = H * W;
  const int tid = threadIdx.x;
  const int lane = tid & 31;
  const int wid = tid >> 5;                         // each wave: 16 rows of M
  const int tileM = blockIdx.x * 128;
  const int b = blockIdx.z;
  const size_t bstr = (size_t)NPIX_TOT * CIN;
  const u16b* yb = y1 + (size_t)b * bstr;

  v8f acc;
  for (int e = 0; e < 8; ++e) acc[e] = 0.0f;

  const int r = tid >> 1, half = tid & 1;
  const int m = tileM + r;
  const int py = (m < HW) ? (m / W) : 0;
  const int px = (m < HW) ? (m % W) : 0;

  for (int kt = 0; kt < 72; ++kt) {
    const int k0 = kt * 32;
    const int tap = k0 >> 8;
    const int c0 = (k0 & 255) + half * 16;
    const int dy = tap / 3 - 1, dx = tap % 3 - 1;
    const int yy = py + dy, xx = px + dx;

    uint4 a0, a1;
    if (m < HW && yy >= 0 && yy < H && xx >= 0 && xx < W) {
      const u16b* g = yb + ((size_t)(yy * W + xx)) * CIN + c0;
      a0 = *(const uint4*)g;
      a1 = *(const uint4*)(g + 8);
    } else {
      a0.x = a0.y = a0.z = a0.w = 0u; a1 = a0;
    }
    *(uint4*)&As[r * 32 + half * 16]     = a0;
    *(uint4*)&As[r * 32 + half * 16 + 8] = a1;

    if (tid < 32) {
      const int nr = tid >> 1, h2 = tid & 1;
      const u16b* gw = wB + (size_t)nr * KTOT + k0 + h2 * 16;
      uint4 b0 = *(const uint4*)gw;
      uint4 b1v = *(const uint4*)(gw + 8);
      *(uint4*)&Bs[nr * 32 + h2 * 16]     = b0;
      *(uint4*)&Bs[nr * 32 + h2 * 16 + 8] = b1v;
    }
    __syncthreads();

    const int khalfA = (lane >> 4) * 8;
    const int rowL   = lane & 15;
    Frag af, bfv;
    const int R = wid * 16 + rowL;
    af.q[0] = *(const uint4*)&As[R * 32 + khalfA];
    af.q[1] = *(const uint4*)&As[R * 32 + 16 + khalfA];
    const int kbB = (lane >> 4) * 8;
    bfv.q[0] = *(const uint4*)&Bs[rowL * 32 + kbB];
    bfv.q[1] = *(const uint4*)&Bs[rowL * 32 + kbB + 16];

    acc = __builtin_amdgcn_wmma_f32_16x16x32_bf16(
        false, af.v, false, bfv.v, (short)0, acc, false, false);

    __syncthreads();
  }

  // scatter: ch = q*3 + a ; anchor n = aoff + a*HW + pixel
  const int lhalf = lane >> 4, ch = lane & 15;
  if (ch < 15) {
    const float bv = bias[ch];
    const int q = ch / 3, a = ch % 3;
    for (int v = 0; v < 8; ++v) {
      const int mrow = tileM + wid * 16 + v + lhalf * 8;
      if (mrow < HW) {
        float val = acc[v] + bv;
        const size_t idx = (size_t)b * NANCH + aoff + a * HW + mrow;
        if (q == 4) scores[idx] = val;
        else        deltas[idx * 4 + q] = val;
      }
    }
  }
}

// ----------------------------------------------------------- box decode ----
__global__ void k_deltas(const float* __restrict__ anch, const float* __restrict__ del,
                         float* __restrict__ boxes) {
  int i = blockIdx.x * blockDim.x + threadIdx.x;
  if (i >= BATCH * NANCH) return;
  const float* A = anch + (size_t)i * 4;
  const float* D = del  + (size_t)i * 4;
  float w = A[2] - A[0], h = A[3] - A[1];
  float cx = A[0] + 0.5f * w, cy = A[1] + 0.5f * h;
  float dw = fminf(D[2], SCALE_CLAMPF), dh = fminf(D[3], SCALE_CLAMPF);
  float pcx = D[0] * w + cx, pcy = D[1] * h + cy;
  float pw = __expf(dw) * w, ph = __expf(dh) * h;
  float* o = boxes + (size_t)i * 4;
  o[0] = pcx - 0.5f * pw; o[1] = pcy - 0.5f * ph;
  o[2] = pcx + 0.5f * pw; o[3] = pcy + 0.5f * ph;
}

// ------------------------------------------------- per-batch descending sort
__global__ void k_sort_init(const float* __restrict__ scores,
                            float* __restrict__ key, int* __restrict__ idx) {
  int t = blockIdx.x * blockDim.x + threadIdx.x;
  if (t >= BATCH * SORTN) return;
  int b = t >> 15, i = t & (SORTN - 1);
  key[t] = (i < NANCH) ? scores[b * NANCH + i] : -3.0e38f;
  idx[t] = i;
}

__global__ void k_bitonic(float* __restrict__ key, int* __restrict__ idx, int kk, int jj) {
  int t = blockIdx.x * blockDim.x + threadIdx.x;
  if (t >= BATCH * SORTN) return;
  int b = t >> 15, i = t & (SORTN - 1);
  int l = i ^ jj;
  if (l <= i) return;
  const int base = b << 15;
  float ki = key[base + i], kl = key[base + l];
  bool up = (i & kk) == 0;                       // descending overall
  bool sw = up ? (ki < kl) : (ki > kl);
  if (sw) {
    key[base + i] = kl; key[base + l] = ki;
    int ti = idx[base + i]; idx[base + i] = idx[base + l]; idx[base + l] = ti;
  }
}

// ------------------------------------------------------------------- NMS ----
__global__ __launch_bounds__(1024) void k_nms(
    const float* __restrict__ boxes, const float* __restrict__ skey,
    const int* __restrict__ sidx, float* __restrict__ out) {
  extern __shared__ char smem[];
  float* bxs = (float*)smem;                  // TOPK*4
  float* scs = bxs + TOPK * 4;                // TOPK
  int*  keep = (int*)(scs + TOPK);            // TOPK
  int*  rank = keep + TOPK;                   // TOPK
  const int b = blockIdx.x, tid = threadIdx.x;

  for (int i = tid; i < TOPK; i += 1024) {
    int gi = sidx[b * SORTN + i];
    float x0 = -1.f, y0 = -1.f, x1 = -1.f, y1 = -1.f, s = -1.f;
    if (gi < NANCH) {
      const float* bp = boxes + ((size_t)b * NANCH + gi) * 4;
      x0 = bp[0]; y0 = bp[1]; x1 = bp[2]; y1 = bp[3];
      s = skey[b * SORTN + i];
    }
    bxs[i * 4 + 0] = x0; bxs[i * 4 + 1] = y0;
    bxs[i * 4 + 2] = x1; bxs[i * 4 + 3] = y1;
    scs[i] = s; keep[i] = 1;
  }
  __syncthreads();

  for (int i = 0; i < TOPK - 1; ++i) {
    if (keep[i]) {
      float ax0 = bxs[i * 4], ay0 = bxs[i * 4 + 1], ax1 = bxs[i * 4 + 2], ay1 = bxs[i * 4 + 3];
      float aarea = (ax1 - ax0) * (ay1 - ay0);
      for (int j = i + 1 + tid; j < TOPK; j += 1024) {
        if (!keep[j]) continue;
        float bx0 = bxs[j * 4], by0 = bxs[j * 4 + 1], bx1 = bxs[j * 4 + 2], by1 = bxs[j * 4 + 3];
        float barea = (bx1 - bx0) * (by1 - by0);
        float iw = fminf(ax1, bx1) - fmaxf(ax0, bx0);
        float ih = fminf(ay1, by1) - fmaxf(ay0, by0);
        iw = iw > 0.f ? iw : 0.f; ih = ih > 0.f ? ih : 0.f;
        float inter = iw * ih;
        float iou = inter > 0.f ? inter / (aarea + barea - inter) : 0.f;
        if (iou > IOU_THR) keep[j] = 0;
      }
    }
    __syncthreads();
  }

  if (tid == 0) {                               // stable kept-first ranks
    int rr = 0;
    for (int i = 0; i < TOPK; ++i) if (keep[i])  rank[i] = rr++;
    for (int i = 0; i < TOPK; ++i) if (!keep[i]) rank[i] = rr++;
  }
  __syncthreads();

  for (int i = tid; i < TOPK; i += 1024) {
    float* o = out + ((size_t)b * TOPK + rank[i]) * 5;
    if (keep[i]) {
      o[0] = bxs[i * 4]; o[1] = bxs[i * 4 + 1];
      o[2] = bxs[i * 4 + 2]; o[3] = bxs[i * 4 + 3]; o[4] = scs[i];
    } else {
      o[0] = o[1] = o[2] = o[3] = o[4] = -1.0f;
    }
  }
}

// ------------------------------------------------------------------ host ----
extern "C" void kernel_launch(void* const* d_in, const int* in_sizes, int n_in,
                              void* d_out, int out_size, void* d_ws, size_t ws_size,
                              hipStream_t stream) {
  (void)in_sizes; (void)n_in; (void)out_size; (void)ws_size;
  const float* feats[3] = {(const float*)d_in[0], (const float*)d_in[1], (const float*)d_in[2]};
  const float* anchors = (const float*)d_in[3];
  const float *w1[3], *b1[3], *w2[3], *b2[3];
  for (int s = 0; s < 3; ++s) {
    w1[s] = (const float*)d_in[4 + 4 * s];
    b1[s] = (const float*)d_in[5 + 4 * s];
    w2[s] = (const float*)d_in[6 + 4 * s];
    b2[s] = (const float*)d_in[7 + 4 * s];
  }
  static const int Hs[3] = {80, 40, 20}, Ws[3] = {80, 40, 20}, HWs[3] = {6400, 1600, 400};
  static const int pixoff[3] = {0, 6400, 8000}, aoff[3] = {0, 19200, 24000};

  char* ws = (char*)d_ws; size_t off = 0;
  auto alloc = [&](size_t bytes) -> void* {
    off = (off + 255) & ~(size_t)255;
    void* p = ws + off; off += bytes; return p;
  };
  u16b* featbf = (u16b*)alloc((size_t)BATCH * NPIX_TOT * CIN * 2);
  u16b* y1bf   = (u16b*)alloc((size_t)BATCH * NPIX_TOT * CIN * 2);
  u16b *wB1[3], *wB2[3];
  for (int s = 0; s < 3; ++s) {
    wB1[s] = (u16b*)alloc((size_t)256 * KTOT * 2);
    wB2[s] = (u16b*)alloc((size_t)16 * KTOT * 2);
  }
  u16b* zp      = (u16b*)alloc(256);
  float* scores = (float*)alloc((size_t)BATCH * NANCH * 4);
  float* deltas = (float*)alloc((size_t)BATCH * NANCH * 4 * 4);
  float* boxes  = (float*)alloc((size_t)BATCH * NANCH * 4 * 4);
  float* skey   = (float*)alloc((size_t)BATCH * SORTN * 4);
  int*   sidx   = (int*)alloc((size_t)BATCH * SORTN * 4);

  k_zero16<<<1, 128, 0, stream>>>(zp, 128);
  for (int s = 0; s < 3; ++s) {
    int total = BATCH * HWs[s] * CIN;
    k_nhwc_bf16<<<(total + 255) / 256, 256, 0, stream>>>(feats[s], featbf, HWs[s], pixoff[s]);
    k_prep_w<<<(256 * KTOT + 255) / 256, 256, 0, stream>>>(w1[s], wB1[s], 256, 256);
    k_prep_w<<<(16 * KTOT + 255) / 256, 256, 0, stream>>>(w2[s], wB2[s], 16, 15);
  }
  for (int s = 0; s < 3; ++s) {
    dim3 g1((HWs[s] + 127) / 128, 2, BATCH);
    k_conv1<<<g1, 256, 0, stream>>>(featbf + (size_t)pixoff[s] * CIN, wB1[s], b1[s],
                                    y1bf + (size_t)pixoff[s] * CIN, zp, Hs[s], Ws[s]);
    dim3 g2((HWs[s] + 127) / 128, 1, BATCH);
    k_conv2<<<g2, 256, 0, stream>>>(y1bf + (size_t)pixoff[s] * CIN, wB2[s], b2[s],
                                    scores, deltas, Hs[s], Ws[s], aoff[s]);
  }
  k_deltas<<<(BATCH * NANCH + 255) / 256, 256, 0, stream>>>(anchors, deltas, boxes);
  k_sort_init<<<(BATCH * SORTN + 255) / 256, 256, 0, stream>>>(scores, skey, sidx);
  for (int k = 2; k <= SORTN; k <<= 1)
    for (int j = k >> 1; j > 0; j >>= 1)
      k_bitonic<<<(BATCH * SORTN + 255) / 256, 256, 0, stream>>>(skey, sidx, k, j);
  k_nms<<<BATCH, 1024, 84000, stream>>>(boxes, skey, sidx, (float*)d_out);
}
